// LiteMLAFix_74517682586441
// MI455X (gfx1250) — matrix operand order repaired and checked by reading, not compiled, not run on hardware
//
// CDNA5 (gfx1250) implementation of the LiteMLA-style block.
// All dense projections run as bf16 V_WMMA_F32_16X16X32_BF16 GEMMs with fp32
// accumulation; each wave owns a 64x32 output tile (4x2 fragments -> 8 WMMAs
// per 32-deep k-step on 6 fragment loads, ~21 FLOP/B from L2). Softmax
// attention / depthwise convs / linear attention stay in fp32 VALU with
// dword-packed bf16 access. Everything is L2-resident (192 MB).

#include <hip/hip_runtime.h>
#include <hip/hip_bf16.h>
#include <math.h>

typedef __attribute__((ext_vector_type(16))) __bf16 v16bf;
typedef __attribute__((ext_vector_type(8)))  float  v8f;

#define DEVINL __device__ __forceinline__

#define BATCH 8
#define HW    1024
#define NTOK  8192      // 8 * 32 * 32 tokens
#define CIN   256
#define CQ    768       // qkv channels (3*HEADS*DIM)
#define CQ3   2304      // ba qkv channels

DEVINL unsigned short f2bf(float f){
  unsigned u = __float_as_uint(f);
  unsigned r = u + 0x7FFFu + ((u >> 16) & 1u);   // round-to-nearest-even
  return (unsigned short)(r >> 16);
}
DEVINL float bf2f(unsigned short h){
  return __uint_as_float(((unsigned)h) << 16);
}
DEVINL float bf2f_lo(unsigned u){ return __uint_as_float(u << 16); }
DEVINL float bf2f_hi(unsigned u){ return __uint_as_float(u & 0xFFFF0000u); }
DEVINL unsigned pack_bf(float a, float b){
  return (unsigned)f2bf(a) | ((unsigned)f2bf(b) << 16);
}

// ---------------------------------------------------------------- converts
__global__ void k_conv_x(const float* __restrict__ x, unsigned short* __restrict__ Xm){
  int gid = blockIdx.x*blockDim.x + threadIdx.x;
  if (gid >= BATCH*CIN*HW) return;
  int hw = gid & 1023;
  int c  = (gid >> 10) & 255;
  int b  = gid >> 18;
  Xm[((size_t)((b << 10) + hw))*CIN + c] = f2bf(x[gid]);  // token-major [8192,256]
}

__global__ void k_conv_flat(const float* __restrict__ in, unsigned short* __restrict__ out, int n){
  int gid = blockIdx.x*blockDim.x + threadIdx.x;
  if (gid < n) out[gid] = f2bf(in[gid]);
}

// ------------------------------------------------- bf16 WMMA GEMM (generic)
// D[M,N] = A[M,K] @ B^T  (+bias), A row-major bf16, B stored [N,K] bf16.
// Wave computes a 64x32 tile (4x2 fragments). Fragment layouts used here (contiguous k, same for A and B):
//   A 16x32 bf16: lane row=l16, VGPR v holds K = 16*half + 2v, 2v+1
//   B 32x16 bf16: lane col=l16, VGPR v holds K = 16*half + 2v, 2v+1
//   D f32 16x16 : lane col=l16, VGPR v -> row = v + 8*half
__global__ void __launch_bounds__(256)
k_gemm_bf16(const unsigned short* __restrict__ A,
            const unsigned short* __restrict__ Bw,
            const float* __restrict__ bias,
            unsigned short* __restrict__ outb,           // bf16 [M,N] (or null)
            float* __restrict__ outf,                    // f32 bohw + BN (or null)
            const float* __restrict__ bng, const float* __restrict__ bnb,
            const float* __restrict__ bnm, const float* __restrict__ bnv,
            int M, int N, int K)
{
  const int lane = threadIdx.x & 31;
  const int wave = threadIdx.x >> 5;
  const int nt32 = N >> 5;
  const int mt64 = M >> 6;
  const int tile = blockIdx.x * (blockDim.x >> 5) + wave;
  if (tile >= mt64 * nt32) return;
  const int tm = tile / nt32, tn = tile - tm*nt32;
  const int row0 = tm << 6, col0 = tn << 5;
  const int half = lane >> 4, l16 = lane & 15;

  const unsigned short* Ar[4];
  #pragma unroll
  for (int i = 0; i < 4; ++i) Ar[i] = A + (size_t)(row0 + 16*i + l16) * K;
  const unsigned short* Bc[2];
  #pragma unroll
  for (int j = 0; j < 2; ++j) Bc[j] = Bw + (size_t)(col0 + 16*j + l16) * K;

  int kposA[8];
  #pragma unroll
  for (int i = 0; i < 8; ++i) kposA[i] = 2 * ((i & 3) + 4*half + ((i >> 2) << 3));

  v8f acc[4][2];
  #pragma unroll
  for (int i = 0; i < 4; ++i)
    #pragma unroll
    for (int j = 0; j < 2; ++j) acc[i][j] = (v8f){};

  union Frag { v16bf v; unsigned u[8]; };

  for (int k0 = 0; k0 < K; k0 += 32){
    if (k0 + 32 < K){                       // gfx1250 global_prefetch_b8 (near)
      __builtin_prefetch(Ar[0] + k0 + 32, 0, 3);
      __builtin_prefetch(Ar[2] + k0 + 32, 0, 3);
      __builtin_prefetch(Bc[0] + k0 + 32, 0, 3);
    }
    Frag a[4], b[2];
    #pragma unroll
    for (int j = 0; j < 2; ++j)
      #pragma unroll
      for (int r = 0; r < 8; ++r)
        b[j].u[r] = *(const unsigned*)(Bc[j] + k0 + 16*half + 2*r);
    #pragma unroll
    for (int i = 0; i < 4; ++i)
      #pragma unroll
      for (int r = 0; r < 8; ++r)
        a[i].u[r] = *(const unsigned*)(Ar[i] + k0 + 16*half + 2*r);

    #pragma unroll
    for (int i = 0; i < 4; ++i){
      acc[i][0] = __builtin_amdgcn_wmma_f32_16x16x32_bf16(false, a[i].v, false, b[0].v,
                                                          (short)0, acc[i][0], false, false);
      acc[i][1] = __builtin_amdgcn_wmma_f32_16x16x32_bf16(false, a[i].v, false, b[1].v,
                                                          (short)0, acc[i][1], false, false);
    }
  }

  #pragma unroll
  for (int tj = 0; tj < 2; ++tj){
    const int n = col0 + tj*16 + l16;
    const float bs = bias ? bias[n] : 0.f;
    float sc = 0.f, sh = 0.f;
    if (outf){ sc = bng[n] * rsqrtf(bnv[n] + 1e-5f); sh = bnb[n] - bnm[n]*sc; }
    #pragma unroll
    for (int ti = 0; ti < 4; ++ti){
      #pragma unroll
      for (int v = 0; v < 8; ++v){
        const int mrow = row0 + ti*16 + v + 8*half;
        const float val = acc[ti][tj][v] + bs;
        if (outb) outb[(size_t)mrow * N + n] = f2bf(val);
        if (outf){
          const int b = mrow >> 10, hw = mrow & 1023;
          outf[(((size_t)b * N + n) << 10) + hw] = val*sc + sh;   // bohw + BN
        }
      }
    }
  }
}

// ------------------------------------------------- window means (q_win,k_win)
// one thread per (b, window, channel-pair); dword-packed bf16 reads
__global__ void k_winmeans(const unsigned short* __restrict__ q2b,
                           float* __restrict__ qwin, float* __restrict__ kwin){
  int gid = blockIdx.x*blockDim.x + threadIdx.x;
  if (gid >= BATCH*16*(CQ/2)) return;
  int cp = gid % 384;
  int p  = (gid / 384) & 15;
  int b  = gid / (384*16);
  int c  = cp*2;
  int h0 = (p >> 2) << 3, w0 = (p & 3) << 3;
  float sq0 = 0.f, sq1 = 0.f, sk0 = 0.f, sk1 = 0.f;
  for (int ih = 0; ih < 8; ++ih)
    for (int iw = 0; iw < 8; ++iw){
      size_t tok = (size_t)(b << 10) + (h0+ih)*32 + (w0+iw);
      const unsigned short* r = q2b + tok*CQ3;
      unsigned uq = *(const unsigned*)(r + c);
      unsigned uk = *(const unsigned*)(r + 768 + c);
      sq0 += bf2f_lo(uq); sq1 += bf2f_hi(uq);
      sk0 += bf2f_lo(uk); sk1 += bf2f_hi(uk);
    }
  size_t o = ((size_t)b*16 + p)*768 + c;
  qwin[o] = sq0*(1.f/64.f); qwin[o+1] = sq1*(1.f/64.f);
  kwin[o] = sk0*(1.f/64.f); kwin[o+1] = sk1*(1.f/64.f);
}

// ------------------------------------------------- 16x16 logits + top-4
__global__ void k_topk(const float* __restrict__ qwin, const float* __restrict__ kwin,
                       int* __restrict__ idx){
  __shared__ float lg[16][16];
  const int b = blockIdx.x;
  const int p = threadIdx.x >> 4, j = threadIdx.x & 15;
  const float scale = rsqrtf(768.f);
  const float* qr = qwin + ((size_t)b*16 + p)*768;
  const float* kr = kwin + ((size_t)b*16 + j)*768;
  float s = 0.f;
  for (int c = 0; c < 768; ++c) s += qr[c]*kr[c];
  lg[p][j] = s * scale;
  __syncthreads();
  if (threadIdx.x < 16){
    const int pp = threadIdx.x;
    float row[16];
    #pragma unroll
    for (int q = 0; q < 16; ++q) row[q] = lg[pp][q];
    for (int t = 0; t < 4; ++t){             // strict > keeps lowest index on tie
      int am = 0; float mv = row[0];
      #pragma unroll
      for (int q = 1; q < 16; ++q) if (row[q] > mv){ mv = row[q]; am = q; }
      idx[((size_t)b*16 + pp)*4 + t] = am;
      row[am] = -3.0e38f;
    }
  }
}

// ------------------------------------------------- LePE 3x3 depthwise (fp32)
// one thread per (token, channel-pair)
__global__ void k_lepe(const unsigned short* __restrict__ q2b, const float* __restrict__ lw,
                       float* __restrict__ lepe_s){
  int gid = blockIdx.x*blockDim.x + threadIdx.x;
  if (gid >= NTOK*(CQ/2)) return;
  int cp = gid % 384;
  int tok = gid / 384;
  int ch = cp*2;
  int b = tok >> 10, hw = tok & 1023, h = hw >> 5, w = hw & 31;
  float s0 = 0.f, s1 = 0.f;
  #pragma unroll
  for (int dy = 0; dy < 3; ++dy){
    int hh = h + dy - 1;
    if ((unsigned)hh >= 32u) continue;
    #pragma unroll
    for (int dx = 0; dx < 3; ++dx){
      int ww = w + dx - 1;
      if ((unsigned)ww >= 32u) continue;
      size_t t2 = (size_t)(b << 10) + hh*32 + ww;
      unsigned u = *(const unsigned*)(q2b + t2*CQ3 + 1536 + ch);
      s0 += bf2f_lo(u) * lw[ch*9 + dy*3 + dx];
      s1 += bf2f_hi(u) * lw[(ch+1)*9 + dy*3 + dx];
    }
  }
  size_t o = (size_t)tok*CQ + ch;
  lepe_s[o] = s0; lepe_s[o+1] = s1;
}

// --------------------------- gathered softmax attention (online), + LePE add
// block = 64 threads (one per query slot), grid = B*16*32 (b, window, head)
__global__ void __launch_bounds__(64)
k_attn(const unsigned short* __restrict__ q2b, const int* __restrict__ idx,
       const float* __restrict__ lepe_s, unsigned short* __restrict__ outb){
  const int bid = blockIdx.x;
  const int m = bid & 31, p = (bid >> 5) & 15, b = bid >> 9;
  const int r = threadIdx.x;
  const int qh = ((p >> 2) << 3) + (r >> 3);
  const int qw = ((p & 3) << 3) + (r & 7);
  const size_t qtok = (size_t)(b << 10) + qh*32 + qw;
  const float scale = rsqrtf(768.f);

  float q[24], accv[24];
  const unsigned* qp = (const unsigned*)(q2b + qtok*CQ3 + m*24);
  #pragma unroll
  for (int j = 0; j < 12; ++j){
    unsigned u = qp[j];
    q[2*j]   = bf2f_lo(u) * scale;
    q[2*j+1] = bf2f_hi(u) * scale;
    accv[2*j] = 0.f; accv[2*j+1] = 0.f;
  }

  float mx = -3.0e38f, l = 0.f;
  for (int t = 0; t < 4; ++t){
    const int ws = idx[((size_t)b*16 + p)*4 + t];
    const int h0 = (ws >> 2) << 3, w0 = (ws & 3) << 3;
    for (int s = 0; s < 64; ++s){
      const size_t kt = (size_t)(b << 10) + (h0 + (s >> 3))*32 + (w0 + (s & 7));
      const unsigned* kp = (const unsigned*)(q2b + kt*CQ3 + 768 + m*24);
      const unsigned* vp = kp + 384;   // +768 bf16 elements
      float dot = 0.f;
      #pragma unroll
      for (int j = 0; j < 12; ++j){
        unsigned u = kp[j];
        dot += q[2*j]*bf2f_lo(u) + q[2*j+1]*bf2f_hi(u);
      }
      if (dot > mx){
        const float corr = __expf(mx - dot);
        l = l*corr + 1.f;
        #pragma unroll
        for (int j = 0; j < 12; ++j){
          unsigned u = vp[j];
          accv[2*j]   = accv[2*j]*corr   + bf2f_lo(u);
          accv[2*j+1] = accv[2*j+1]*corr + bf2f_hi(u);
        }
        mx = dot;
      } else {
        const float w = __expf(dot - mx);
        l += w;
        #pragma unroll
        for (int j = 0; j < 12; ++j){
          unsigned u = vp[j];
          accv[2*j]   += w*bf2f_lo(u);
          accv[2*j+1] += w*bf2f_hi(u);
        }
      }
    }
  }
  const float inv = 1.f / l;
  const size_t obase = qtok*CQ + (size_t)m*24;
  unsigned* o32 = (unsigned*)(outb + obase);
  const float* lp = lepe_s + obase;
  #pragma unroll
  for (int j = 0; j < 12; ++j)
    o32[j] = pack_bf(accv[2*j]*inv + lp[2*j], accv[2*j+1]*inv + lp[2*j+1]);
}

// ------------------------------------------------- ag: depthwise 5x5 pad 2
// one thread per (token, channel-pair)
__global__ void k_ag_dw(const unsigned short* __restrict__ qkv_tb, const float* __restrict__ dww,
                        unsigned short* __restrict__ ag1b){
  int gid = blockIdx.x*blockDim.x + threadIdx.x;
  if (gid >= NTOK*(CQ/2)) return;
  int cp = gid % 384;
  int tok = gid / 384;
  int ch = cp*2;
  int b = tok >> 10, hw = tok & 1023, h = hw >> 5, w = hw & 31;
  float s0 = 0.f, s1 = 0.f;
  for (int dy = 0; dy < 5; ++dy){
    int hh = h + dy - 2;
    if ((unsigned)hh >= 32u) continue;
    for (int dx = 0; dx < 5; ++dx){
      int ww = w + dx - 2;
      if ((unsigned)ww >= 32u) continue;
      size_t t2 = (size_t)(b << 10) + hh*32 + ww;
      unsigned u = *(const unsigned*)(qkv_tb + t2*CQ + ch);
      s0 += bf2f_lo(u) * dww[ch*25 + dy*5 + dx];
      s1 += bf2f_hi(u) * dww[(ch+1)*25 + dy*5 + dx];
    }
  }
  *(unsigned*)(ag1b + (size_t)tok*CQ + ch) = pack_bf(s0, s1);
}

// ------------------------------------------------- ag: grouped 1x1 (groups=96)
// one thread per (token, output-channel-pair); pair never crosses a group of 8
__global__ void k_ag_pw(const unsigned short* __restrict__ ag1b, const float* __restrict__ pww,
                        unsigned short* __restrict__ ag2b){
  int gid = blockIdx.x*blockDim.x + threadIdx.x;
  if (gid >= NTOK*(CQ/2)) return;
  int op = (gid % 384)*2;
  int tok = gid / 384;
  int g = op >> 3;
  const unsigned* rr = (const unsigned*)(ag1b + (size_t)tok*CQ + g*8);
  float in[8];
  #pragma unroll
  for (int i = 0; i < 4; ++i){
    unsigned u = rr[i];
    in[2*i] = bf2f_lo(u); in[2*i+1] = bf2f_hi(u);
  }
  float s0 = 0.f, s1 = 0.f;
  #pragma unroll
  for (int i = 0; i < 8; ++i){
    s0 += in[i] * pww[op*8 + i];
    s1 += in[i] * pww[(op+1)*8 + i];
  }
  *(unsigned*)(ag2b + (size_t)tok*CQ + op) = pack_bf(s0, s1);
}

// ------------------------------------------------- ReLU linear attention
// block = (b, head); 256 threads; kv = relu(k)^T [v|1] reduced over 1024 tokens
__global__ void __launch_bounds__(256)
k_linatt(const unsigned short* __restrict__ inb, unsigned short* __restrict__ catb, int coff){
  __shared__ float red[72*8];
  __shared__ float kvs[72];
  const int b = blockIdx.x >> 5;
  const int head = blockIdx.x & 31;
  const int tid = threadIdx.x;

  float kvp[72];
  #pragma unroll
  for (int j = 0; j < 72; ++j) kvp[j] = 0.f;

  for (int t = 0; t < 4; ++t){
    const size_t tok = (size_t)(b << 10) + tid + t*256;
    const unsigned* xp = (const unsigned*)(inb + tok*CQ + head*24);
    float kk[8], vv[9];
    #pragma unroll
    for (int i = 0; i < 4; ++i){
      unsigned u = xp[4 + i];          // k channels 8..15
      kk[2*i]   = fmaxf(bf2f_lo(u), 0.f);
      kk[2*i+1] = fmaxf(bf2f_hi(u), 0.f);
      unsigned v = xp[8 + i];          // v channels 16..23
      vv[2*i]   = bf2f_lo(v);
      vv[2*i+1] = bf2f_hi(v);
    }
    vv[8] = 1.f;
    #pragma unroll
    for (int d = 0; d < 8; ++d)
      #pragma unroll
      for (int e = 0; e < 9; ++e) kvp[d*9 + e] += kk[d]*vv[e];
  }

  const int lane = tid & 31, wv = tid >> 5;
  #pragma unroll
  for (int j = 0; j < 72; ++j){
    float v = kvp[j];
    for (int off = 16; off > 0; off >>= 1) v += __shfl_down(v, off);
    if (lane == 0) red[j*8 + wv] = v;
  }
  __syncthreads();
  if (tid < 72){
    float s = 0.f;
    #pragma unroll
    for (int w2 = 0; w2 < 8; ++w2) s += red[tid*8 + w2];
    kvs[tid] = s;
  }
  __syncthreads();

  for (int t = 0; t < 4; ++t){
    const size_t tok = (size_t)(b << 10) + tid + t*256;
    const unsigned* xp = (const unsigned*)(inb + tok*CQ + head*24);
    float qv[8];
    #pragma unroll
    for (int i = 0; i < 4; ++i){
      unsigned u = xp[i];              // q channels 0..7
      qv[2*i]   = fmaxf(bf2f_lo(u), 0.f);
      qv[2*i+1] = fmaxf(bf2f_hi(u), 0.f);
    }
    float o9[9];
    #pragma unroll
    for (int e = 0; e < 9; ++e){
      float s = 0.f;
      #pragma unroll
      for (int d = 0; d < 8; ++d) s += qv[d]*kvs[d*9 + e];
      o9[e] = s;
    }
    const float inv = 1.f/(o9[8] + 1e-15f);
    unsigned* op = (unsigned*)(catb + tok*512 + coff + head*8);
    #pragma unroll
    for (int j = 0; j < 4; ++j)
      op[j] = pack_bf(o9[2*j]*inv, o9[2*j+1]*inv);
  }
}

// ================================================================= launcher
extern "C" void kernel_launch(void* const* d_in, const int* in_sizes, int n_in,
                              void* d_out, int out_size, void* d_ws, size_t ws_size,
                              hipStream_t stream){
  (void)in_sizes; (void)n_in; (void)out_size; (void)ws_size;
  const float* x         = (const float*)d_in[0];
  const float* qkv_w     = (const float*)d_in[1];
  const float* ag_dw_w   = (const float*)d_in[2];
  const float* ag_pw_w   = (const float*)d_in[3];
  const float* ba_qkv_w  = (const float*)d_in[4];
  const float* ba_qkv_b  = (const float*)d_in[5];
  const float* ba_wo_w   = (const float*)d_in[6];
  const float* ba_wo_b   = (const float*)d_in[7];
  const float* ba_lepe_w = (const float*)d_in[8];
  const float* proj_w    = (const float*)d_in[9];
  const float* bn_g      = (const float*)d_in[10];
  const float* bn_b      = (const float*)d_in[11];
  const float* bn_m      = (const float*)d_in[12];
  const float* bn_v      = (const float*)d_in[13];
  float* out = (float*)d_out;

  char* ws = (char*)d_ws;
  size_t off = 0;
  auto alloc = [&](size_t bytes)->char*{
    char* p = ws + off;
    off = (off + bytes + 255) & ~(size_t)255;
    return p;
  };
  unsigned short* Xmat   = (unsigned short*)alloc((size_t)NTOK*CIN*2);
  unsigned short* Wqkv   = (unsigned short*)alloc((size_t)CQ*CIN*2);
  unsigned short* Wba    = (unsigned short*)alloc((size_t)CQ3*CQ*2);
  unsigned short* Wwo    = (unsigned short*)alloc((size_t)CQ*CQ*2);
  unsigned short* Wproj  = (unsigned short*)alloc((size_t)CIN*512*2);
  unsigned short* qkv_tb = (unsigned short*)alloc((size_t)NTOK*CQ*2);
  unsigned short* q2b    = (unsigned short*)alloc((size_t)NTOK*CQ3*2);
  unsigned short* attlep = (unsigned short*)alloc((size_t)NTOK*CQ*2);
  unsigned short* baoutb = (unsigned short*)alloc((size_t)NTOK*CQ*2);
  unsigned short* ag1b   = (unsigned short*)alloc((size_t)NTOK*CQ*2);
  unsigned short* ag2b   = (unsigned short*)alloc((size_t)NTOK*CQ*2);
  unsigned short* catb   = (unsigned short*)alloc((size_t)NTOK*512*2);
  float*          lepe_s = (float*)alloc((size_t)NTOK*CQ*4);
  float*          qwin   = (float*)alloc((size_t)BATCH*16*CQ*4);
  float*          kwin   = (float*)alloc((size_t)BATCH*16*CQ*4);
  int*            idxp   = (int*)alloc((size_t)BATCH*16*4*4);

  int n;
  n = BATCH*CIN*HW; k_conv_x<<<(n+255)/256, 256, 0, stream>>>(x, Xmat);
  n = CQ*CIN;       k_conv_flat<<<(n+255)/256, 256, 0, stream>>>(qkv_w,    Wqkv,  n);
  n = CQ3*CQ;       k_conv_flat<<<(n+255)/256, 256, 0, stream>>>(ba_qkv_w, Wba,   n);
  n = CQ*CQ;        k_conv_flat<<<(n+255)/256, 256, 0, stream>>>(ba_wo_w,  Wwo,   n);
  n = CIN*512;      k_conv_flat<<<(n+255)/256, 256, 0, stream>>>(proj_w,   Wproj, n);

  { // GEMM1: qkv = x @ qkv_w^T   [8192,256]x[768,256]
    int M = NTOK, N = CQ, K = CIN;
    int tiles = (M >> 6)*(N >> 5);
    k_gemm_bf16<<<(tiles+7)/8, 256, 0, stream>>>(Xmat, Wqkv, nullptr, qkv_tb, nullptr,
                                                 nullptr, nullptr, nullptr, nullptr, M, N, K);
  }
  { // GEMM2: q2 = qkv @ ba_qkv_w^T + b   [8192,768]x[2304,768]
    int M = NTOK, N = CQ3, K = CQ;
    int tiles = (M >> 6)*(N >> 5);
    k_gemm_bf16<<<(tiles+7)/8, 256, 0, stream>>>(qkv_tb, Wba, ba_qkv_b, q2b, nullptr,
                                                 nullptr, nullptr, nullptr, nullptr, M, N, K);
  }
  n = BATCH*16*(CQ/2); k_winmeans<<<(n+255)/256, 256, 0, stream>>>(q2b, qwin, kwin);
  k_topk<<<BATCH, 256, 0, stream>>>(qwin, kwin, idxp);
  n = NTOK*(CQ/2);     k_lepe<<<(n+255)/256, 256, 0, stream>>>(q2b, ba_lepe_w, lepe_s);
  k_attn<<<BATCH*16*32, 64, 0, stream>>>(q2b, idxp, lepe_s, attlep);
  { // GEMM3: ba_out = (attn+lepe) @ wo^T + b   [8192,768]x[768,768]
    int M = NTOK, N = CQ, K = CQ;
    int tiles = (M >> 6)*(N >> 5);
    k_gemm_bf16<<<(tiles+7)/8, 256, 0, stream>>>(attlep, Wwo, ba_wo_b, baoutb, nullptr,
                                                 nullptr, nullptr, nullptr, nullptr, M, N, K);
  }
  n = NTOK*(CQ/2);  k_ag_dw<<<(n+255)/256, 256, 0, stream>>>(qkv_tb, ag_dw_w, ag1b);
  n = NTOK*(CQ/2);  k_ag_pw<<<(n+255)/256, 256, 0, stream>>>(ag1b, ag_pw_w, ag2b);
  k_linatt<<<BATCH*32, 256, 0, stream>>>(baoutb, catb, 0);
  k_linatt<<<BATCH*32, 256, 0, stream>>>(ag2b,   catb, 256);
  { // GEMM4: out = cat @ proj^T, fused BN, scatter to bohw f32
    int M = NTOK, N = CIN, K = 512;
    int tiles = (M >> 6)*(N >> 5);
    k_gemm_bf16<<<(tiles+7)/8, 256, 0, stream>>>(catb, Wproj, nullptr, nullptr, out,
                                                 bn_g, bn_b, bn_m, bn_v, M, N, K);
  }
}
